// CriticBaseline_39221641347076
// MI455X (gfx1250) — matrix operand order repaired and checked by reading, not compiled, not run on hardware
//
#include <hip/hip_runtime.h>
#include <hip/hip_bf16.h>

typedef __bf16 bf16;
typedef __attribute__((ext_vector_type(16))) __bf16 v16bf;
typedef __attribute__((ext_vector_type(8)))  float  v8f;

#define GAMMA_F   0.99f
#define GAMMA64_F 0.52559647f   // 0.99^64

constexpr int T    = 65536;
constexpr int DH   = 1024;          // D == H == 1024
constexpr int ROWS = 32;            // rows per block
constexpr int LDH  = 1032;          // LDS row stride in bf16 (bank-conflict padding)
constexpr int TILE_BYTES = ROWS * LDH * 2;            // 66048
constexpr int SMEM_BYTES = 2 * TILE_BYTES + ROWS * 4; // X + Y ping-pong + Vred

union BFrag { int4 q[2]; v16bf v; };
union Pack4 { bf16 h[4]; int2 v; };

// ---------------------------------------------------------------------------
// Kernel 1: discounted returns (reverse scan), single 1024-thread block.
// G[i] = r[i] + gamma * G[i+1]
// ---------------------------------------------------------------------------
__global__ __launch_bounds__(1024) void returns_kernel(
    const float* __restrict__ r, float* __restrict__ G) {
  __shared__ float carry[1024];
  __shared__ float excl[1024];
  const int t = threadIdx.x;
  const int base = t * 64;

  float c = 0.f;
  #pragma unroll 4
  for (int k = 63; k >= 0; --k) c = r[base + k] + GAMMA_F * c;
  carry[t] = c;
  __syncthreads();

  if (t == 0) {
    float e = 0.f;
    excl[1023] = 0.f;
    for (int b = 1023; b >= 1; --b) {
      e = carry[b] + GAMMA64_F * e;
      excl[b - 1] = e;
    }
  }
  __syncthreads();

  float g = excl[t];
  for (int k = 63; k >= 0; --k) {
    g = r[base + k] + GAMMA_F * g;
    G[base + k] = g;
  }
}

// ---------------------------------------------------------------------------
// Kernel 2: convert fp32 weights [1024x1024] row-major -> bf16 swizzled into
// WMMA B-fragment order. Tile (kt,nt) = 32x16; within tile, half index
// lane*16 + v*2 + h  maps to  (k = (v>=4?16:0)+(lane>=16?8:0)+2(v&3)+h, n = lane%16)
// per CDNA5 ISA 7.12.2 16-bit B layout. Each lane then loads its fragment
// as two contiguous b128s.
// ---------------------------------------------------------------------------
__global__ __launch_bounds__(256) void swizzle_kernel(
    const float* __restrict__ W, bf16* __restrict__ Ws) {
  int idx  = blockIdx.x * 256 + threadIdx.x;   // < 1024*1024
  int tile = idx >> 9;                         // /512 halves per tile
  int rr   = idx & 511;
  int lane = rr >> 4;                          // 0..31
  int v    = (rr >> 1) & 7;                    // VGPR 0..7
  int h    = rr & 1;                           // lo/hi half
  int kt   = tile >> 6;                        // K block (32 wide)
  int nt   = tile & 63;                        // N block (16 wide)
  int n    = lane & 15;
  int k    = ((v >> 2) << 4) + ((lane >> 4) << 3) + (v & 3) * 2 + h;
  Ws[idx] = (bf16)W[(size_t)(kt * 32 + k) * DH + nt * 16 + n];
}

// ---------------------------------------------------------------------------
// Fused MLP block kernel
// ---------------------------------------------------------------------------
__device__ __forceinline__ v16bf load_a_frag(const bf16* Xs, int strip, int kk,
                                             int lane) {
  // A 16x32 bf16 layout (ISA 7.12.2): lane%16 = row M; lanes<16 hold K 0..7
  // (V0-V3) and 16..23 (V4-V7); lanes>=16 hold K 8..15 and 24..31.
  const int lrow = lane & 15;
  const int ksel = (lane & 16) ? 8 : 0;
  const char* p = (const char*)(Xs + (strip * 16 + lrow) * LDH + kk * 32 + ksel);
  BFrag f;
  f.q[0] = *(const int4*)p;          // K base .. base+7
  f.q[1] = *(const int4*)(p + 32);   // K base+16 .. base+23
  return f.v;
}

__device__ __forceinline__ v16bf load_b_frag(const bf16* Ws, int kk, int nt,
                                             int lane) {
  const int4* p = (const int4*)(Ws + (((size_t)kk * 64 + nt) << 9) + (lane << 4));
  BFrag f;
  f.q[0] = p[0];
  f.q[1] = p[1];
  return f.v;
}

__device__ __forceinline__ void gemm_relu_layer(const bf16* Xs, bf16* Ys,
                                                const bf16* __restrict__ Ws,
                                                const float* __restrict__ bias) {
  const int lane = threadIdx.x & 31;
  const int wave = threadIdx.x >> 5;   // 0..7
  const int ntBase = wave * 8;         // 8 col tiles (128 cols) per wave

  for (int g = 0; g < 2; ++g) {        // two groups of 4 col tiles
    const int nt0 = ntBase + g * 4;
    v8f acc[2][4];
    #pragma unroll
    for (int s = 0; s < 2; ++s)
      #pragma unroll
      for (int t = 0; t < 4; ++t)
        acc[s][t] = v8f{0.f, 0.f, 0.f, 0.f, 0.f, 0.f, 0.f, 0.f};

    float bl[4];
    #pragma unroll
    for (int t = 0; t < 4; ++t)
      bl[t] = bias[(nt0 + t) * 16 + (lane & 15)];

    for (int kk = 0; kk < DH / 32; ++kk) {
      v16bf a0 = load_a_frag(Xs, 0, kk, lane);
      v16bf a1 = load_a_frag(Xs, 1, kk, lane);
      #pragma unroll
      for (int t = 0; t < 4; ++t) {
        v16bf b = load_b_frag(Ws, kk, nt0 + t, lane);
        acc[0][t] = __builtin_amdgcn_wmma_f32_16x16x32_bf16(
            false, a0, false, b, (short)0, acc[0][t], false, false);
        acc[1][t] = __builtin_amdgcn_wmma_f32_16x16x32_bf16(
            false, a1, false, b, (short)0, acc[1][t], false, false);
      }
    }

    // Epilogue: bias + ReLU + convert to bf16 into output tile.
    // C/D layout: VGPR r -> M=r (lanes 0-15) / M=8+r (lanes 16-31), N=lane%16.
    #pragma unroll
    for (int s = 0; s < 2; ++s)
      #pragma unroll
      for (int t = 0; t < 4; ++t) {
        const int col = (nt0 + t) * 16 + (lane & 15);
        #pragma unroll
        for (int rr = 0; rr < 8; ++rr) {
          float v = acc[s][t][rr] + bl[t];
          v = v > 0.f ? v : 0.f;
          const int row = s * 16 + ((lane & 16) ? 8 + rr : rr);
          Ys[row * LDH + col] = (bf16)v;
        }
      }
  }
}

__global__ __launch_bounds__(256) void mlp_kernel(
    const float* __restrict__ obs,
    const bf16*  __restrict__ W1s, const float* __restrict__ b1,
    const bf16*  __restrict__ W2s, const float* __restrict__ b2,
    const float* __restrict__ W3,  const float* __restrict__ b3,
    const float* __restrict__ G,   float* __restrict__ out) {
  extern __shared__ char smem[];
  bf16*  Xs   = (bf16*)smem;
  bf16*  Ys   = (bf16*)(smem + TILE_BYTES);
  float* Vred = (float*)(smem + 2 * TILE_BYTES);

  const int row0 = blockIdx.x * ROWS;

  // Stage obs tile (32 x 1024 fp32) -> bf16 LDS, coalesced float4 loads.
  const float4* obs4 = (const float4*)(obs + (size_t)row0 * DH);
  for (int i = threadIdx.x; i < ROWS * (DH / 4); i += 256) {
    const int row = i >> 8;        // 256 float4 per row
    const int c4  = i & 255;
    float4 f = obs4[i];
    Pack4 p;
    p.h[0] = (bf16)f.x; p.h[1] = (bf16)f.y;
    p.h[2] = (bf16)f.z; p.h[3] = (bf16)f.w;
    *(int2*)(Xs + row * LDH + c4 * 4) = p.v;
  }
  if (threadIdx.x < ROWS) Vred[threadIdx.x] = 0.f;
  __syncthreads();

  gemm_relu_layer(Xs, Ys, W1s, b1);   // h1 = relu(obs @ W1 + b1)
  __syncthreads();
  gemm_relu_layer(Ys, Xs, W2s, b2);   // h2 = relu(h1 @ W2 + b2)  (overwrites obs tile)
  __syncthreads();

  // Layer 3: V = h2 @ W3 + b3 ; out = G - V
  const int row  = threadIdx.x & 31;
  const int part = threadIdx.x >> 5;  // 8 partials of 128 cols each
  const bf16*  h  = Xs + row * LDH + part * 128;
  const float* w3 = W3 + part * 128;
  float s = 0.f;
  #pragma unroll 8
  for (int c = 0; c < 128; ++c) s += (float)h[c] * w3[c];
  atomicAdd(&Vred[row], s);
  __syncthreads();

  if (threadIdx.x < ROWS) {
    const int gr = row0 + threadIdx.x;
    out[gr] = G[gr] - (Vred[threadIdx.x] + b3[0]);
  }
}

// ---------------------------------------------------------------------------
// Host launcher
// ---------------------------------------------------------------------------
extern "C" void kernel_launch(void* const* d_in, const int* in_sizes, int n_in,
                              void* d_out, int out_size, void* d_ws, size_t ws_size,
                              hipStream_t stream) {
  const float* rewards = (const float*)d_in[0];
  const float* obs     = (const float*)d_in[1];
  const float* W1      = (const float*)d_in[2];
  const float* b1      = (const float*)d_in[3];
  const float* W2      = (const float*)d_in[4];
  const float* b2      = (const float*)d_in[5];
  const float* W3      = (const float*)d_in[6];
  const float* b3      = (const float*)d_in[7];
  float* out = (float*)d_out;

  // Workspace layout: G (T fp32) | W1 swizzled bf16 (2MB) | W2 swizzled bf16 (2MB)
  float* Gws = (float*)d_ws;
  bf16*  W1s = (bf16*)((char*)d_ws + (size_t)T * sizeof(float));
  bf16*  W2s = (bf16*)((char*)d_ws + (size_t)T * sizeof(float) +
                       (size_t)DH * DH * sizeof(bf16));

  returns_kernel<<<1, 1024, 0, stream>>>(rewards, Gws);
  swizzle_kernel<<<DH * DH / 256, 256, 0, stream>>>(W1, W1s);
  swizzle_kernel<<<DH * DH / 256, 256, 0, stream>>>(W2, W2s);
  mlp_kernel<<<T / ROWS, 256, SMEM_BYTES, stream>>>(obs, W1s, b1, W2s, b2,
                                                    W3, b3, Gws, out);
}
